// GATNet_2276332667608
// MI455X (gfx1250) — compile-verified, matrix operand order checked by reading, not run on hardware
//
#include <hip/hip_runtime.h>
#include <hip/hip_bf16.h>

// ---------------------------------------------------------------------------
// GAT 2-layer forward for MI455X (gfx1250, wave32, WMMA + async LDS staging).
//   layer: h = X*W (WMMA bf16->f32, A/B tiles staged to LDS with
//          GLOBAL_LOAD_ASYNC_TO_LDS_B128 / ASYNCcnt), alpha = X*(W*a) folded
//          matvec, edge softmax via float atomic max/add, wave-per-edge
//          scatter aggregation with global_atomic_add_f32.
// ---------------------------------------------------------------------------

#define ATT_SLOPE 0.2f
#define ACT_SLOPE 0.01f

typedef __attribute__((ext_vector_type(16))) __bf16 v16bf;
typedef __attribute__((ext_vector_type(8)))  float  v8f;

union Frag16 { v16bf v; unsigned int u[8]; };

#if defined(__gfx1250__) && __has_builtin(__builtin_amdgcn_global_load_async_to_lds_b128)
#define USE_ASYNC_LDS 1
#else
#define USE_ASYNC_LDS 0
#endif

// The async-to-LDS builtins take pointers to 16B int vectors in the global
// (AS1) and LDS (AS3) address spaces (per hipcc's diagnostic).
typedef int v4i_b128 __attribute__((__vector_size__(4 * sizeof(int))));
typedef __attribute__((address_space(1))) v4i_b128* gptr_b128;
typedef __attribute__((address_space(3))) v4i_b128* lptr_b128;

__device__ __forceinline__ gptr_b128 to_as1(const void* p) {
    return (gptr_b128)(unsigned long long)p;
}
__device__ __forceinline__ lptr_b128 to_as3(void* p) {
    return (lptr_b128)(unsigned long long)p;
}

__device__ __forceinline__ void wait_async0() {
#if __has_builtin(__builtin_amdgcn_s_wait_asynccnt)
    __builtin_amdgcn_s_wait_asynccnt(0);
#elif defined(__gfx1250__)
    asm volatile("s_wait_asynccnt 0x0" ::: "memory");
#endif
}

__device__ __forceinline__ unsigned short f32_to_bf16(float f) {
    unsigned int u = __float_as_uint(f);
    unsigned int r = u + 0x7FFFu + ((u >> 16) & 1u);   // round-to-nearest-even
    return (unsigned short)(r >> 16);
}

__device__ __forceinline__ unsigned int pack2_bf16(float f0, float f1) {
    return (unsigned int)f32_to_bf16(f0) | ((unsigned int)f32_to_bf16(f1) << 16);
}

__device__ __forceinline__ float atomicMaxFloat(float* addr, float value) {
    if (value >= 0.0f)
        return __int_as_float(atomicMax((int*)addr, __float_as_int(value)));
    else
        return __uint_as_float(atomicMin((unsigned int*)addr, __float_as_uint(value)));
}

// ---------------------------------------------------------------------------
// Prep (single block): fold attention vectors through W, and emit W^T in bf16.
//   w1s[k] = sum_j W1[k,j]*a1[j]   w1d[k] = sum_j W1[k,j]*a1[128+j]
//   W1t[n*128+k] = bf16(W1[k,n])   (column-major so B-frag K-pairs are packed)
// ---------------------------------------------------------------------------
__global__ __launch_bounds__(256) void gat_prep(
    const float* __restrict__ W1, const float* __restrict__ a1,
    const float* __restrict__ W2, const float* __restrict__ a2,
    float* __restrict__ w1s, float* __restrict__ w1d,
    float* __restrict__ w2s, float* __restrict__ w2d,
    unsigned short* __restrict__ W1t, unsigned short* __restrict__ W2t)
{
    const int t = threadIdx.x;
    if (t < 128) {
        float s = 0.f, d = 0.f;
        for (int j = 0; j < 128; ++j) {
            float w = W1[t * 128 + j];
            s += w * a1[j];
            d += w * a1[128 + j];
        }
        w1s[t] = s; w1d[t] = d;
    } else {
        const int k = t - 128;
        float s = 0.f, d = 0.f;
        for (int j = 0; j < 64; ++j) {
            float w = W2[k * 64 + j];
            s += w * a2[j];
            d += w * a2[64 + j];
        }
        w2s[k] = s; w2d[k] = d;
    }
    for (int i = t; i < 128 * 128; i += 256) {
        int n = i >> 7, k = i & 127;
        W1t[i] = f32_to_bf16(W1[k * 128 + n]);
    }
    for (int i = t; i < 64 * 128; i += 256) {
        int n = i >> 7, k = i & 127;
        W2t[i] = f32_to_bf16(W2[k * 64 + n]);
    }
}

// ---------------------------------------------------------------------------
// Pack f32 rows into bf16-pair (u32) rows: dst[i] = bf16(src[2i],src[2i+1]).
// ---------------------------------------------------------------------------
__global__ __launch_bounds__(256) void pack_bf16_pairs(
    const float* __restrict__ src, unsigned int* __restrict__ dst, size_t npairs)
{
    const size_t i = (size_t)blockIdx.x * blockDim.x + threadIdx.x;
    if (i >= npairs) return;
    dst[i] = pack2_bf16(src[2 * i], src[2 * i + 1]);
}

// ---------------------------------------------------------------------------
// Between-layer activation: leaky_relu in place (f32) + emit packed bf16 copy
// for the next WMMA GEMM (free bf16 conversion pass).
// ---------------------------------------------------------------------------
__global__ __launch_bounds__(256) void leaky_pack_bf16(
    float* __restrict__ p, unsigned int* __restrict__ pb, size_t npairs)
{
    const size_t i = (size_t)blockIdx.x * blockDim.x + threadIdx.x;
    if (i >= npairs) return;
    float v0 = p[2 * i], v1 = p[2 * i + 1];
    v0 = v0 > 0.f ? v0 : ACT_SLOPE * v0;
    v1 = v1 > 0.f ? v1 : ACT_SLOPE * v1;
    p[2 * i] = v0; p[2 * i + 1] = v1;
    pb[i] = pack2_bf16(v0, v1);
}

// ---------------------------------------------------------------------------
// WMMA GEMM:  C[nrows x F] = A[nrows x 128] * W[128 x F].
// A given as packed-bf16 pairs [Npad x 64 u32], W as packed bf16 W^T.
// Block = 128 threads (4 waves), 64 rows/block, 16 rows/wave.
// Tiles staged to LDS via GLOBAL_LOAD_ASYNC_TO_LDS_B128 (ASYNCcnt): each lane
// moves 16B per async op, then s_wait_asynccnt 0 + barrier before WMMA.
// A-frag layout (16-bit A 16x32): lane<16 -> K 0..7 / 16..23, lane>=16 ->
// K 8..15 / 24..31, packed 2 per VGPR.  B symmetric with N = lane&15.
// ---------------------------------------------------------------------------
template<int F>
__global__ __launch_bounds__(128) void gemm_wmma_bf16(
    const unsigned int* __restrict__ Ab, const unsigned int* __restrict__ WtPacked,
    float* __restrict__ C, int nrows)
{
    constexpr int ROWS = 64;
    constexpr int KD2  = 64;                       // 128 K-values / 2 per u32
    __shared__ unsigned int sA[ROWS * KD2];        // 16 KB
    __shared__ unsigned int sB[F * KD2];           // 32 KB (F=128) / 16 KB

    const int tid  = threadIdx.x;
    const int row0 = blockIdx.x * ROWS;
    const unsigned int* gA = Ab + (size_t)row0 * KD2;

#if USE_ASYNC_LDS
    // Async DMA into LDS: 16B per lane per op, tracked by ASYNCcnt.
    for (int i = tid * 4; i < F * KD2; i += 128 * 4)
        __builtin_amdgcn_global_load_async_to_lds_b128(
            to_as1(WtPacked + i), to_as3(&sB[i]), 0, 0);
    for (int i = tid * 4; i < ROWS * KD2; i += 128 * 4)
        __builtin_amdgcn_global_load_async_to_lds_b128(
            to_as1(gA + i), to_as3(&sA[i]), 0, 0);
    wait_async0();
#else
    for (int i = tid; i < F * KD2; i += 128) sB[i] = WtPacked[i];
    for (int i = tid; i < ROWS * KD2; i += 128) sA[i] = gA[i];
#endif
    __syncthreads();

    const int wave = tid >> 5;
    const int lane = tid & 31;
    const int half = lane >> 4;
    const int mrow = lane & 15;
    const int lrow = wave * 16 + mrow;

    constexpr int NT = F / 16;
    const v8f vzero = {0.f, 0.f, 0.f, 0.f, 0.f, 0.f, 0.f, 0.f};
    v8f acc[NT];
#pragma unroll
    for (int t = 0; t < NT; ++t) acc[t] = vzero;

#pragma unroll
    for (int kc = 0; kc < 4; ++kc) {               // K chunks of 32
        Frag16 a;
        const unsigned int* pa = &sA[lrow * KD2 + kc * 16 + half * 4];
#pragma unroll
        for (int v = 0; v < 4; ++v) { a.u[v] = pa[v]; a.u[4 + v] = pa[8 + v]; }
#pragma unroll
        for (int t = 0; t < NT; ++t) {
            Frag16 b;
            const unsigned int* pb =
                &sB[(t * 16 + mrow) * KD2 + kc * 16 + half * 4];
#pragma unroll
            for (int v = 0; v < 4; ++v) { b.u[v] = pb[v]; b.u[4 + v] = pb[8 + v]; }
            acc[t] = __builtin_amdgcn_wmma_f32_16x16x32_bf16(
                false, a.v, false, b.v, (short)0, acc[t], false, false);
        }
    }

    // C/D layout: VGPR r -> M = half*8 + r, N = t*16 + mrow
#pragma unroll
    for (int t = 0; t < NT; ++t) {
#pragma unroll
        for (int r = 0; r < 8; ++r) {
            const int grow = row0 + wave * 16 + half * 8 + r;
            if (grow < nrows) C[(size_t)grow * F + (t * 16 + mrow)] = acc[t][r];
        }
    }
}

// ---------------------------------------------------------------------------
// alpha_{src,dst}[i] = dot(X[i,:], wfold_{s,d})   (wave per node, K=128)
// ---------------------------------------------------------------------------
__global__ __launch_bounds__(256) void alpha_matvec(
    const float* __restrict__ X, const float* __restrict__ ws,
    const float* __restrict__ wd, float* __restrict__ as_,
    float* __restrict__ ad_, int n)
{
    const int gw = (int)((blockIdx.x * blockDim.x + threadIdx.x) >> 5);
    const int lane = threadIdx.x & 31;
    if (gw >= n) return;
    const float* row = X + (size_t)gw * 128;
    float s = 0.f, d = 0.f;
#pragma unroll
    for (int c = lane; c < 128; c += 32) {
        const float v = row[c];
        s += v * ws[c];
        d += v * wd[c];
    }
#pragma unroll
    for (int o = 16; o > 0; o >>= 1) {
        s += __shfl_xor(s, o, 32);
        d += __shfl_xor(d, o, 32);
    }
    if (lane == 0) { as_[gw] = s; ad_[gw] = d; }
}

// ---------------------------------------------------------------------------
// Edge pass 1: segment max of leaky_relu(alpha_s[src]+alpha_d[dst])
// ---------------------------------------------------------------------------
__global__ __launch_bounds__(256) void edge_max(
    const int* __restrict__ ei, const float* __restrict__ as_,
    const float* __restrict__ ad_, float* __restrict__ m, int E)
{
    const int e = blockIdx.x * blockDim.x + threadIdx.x;
    if (e >= E) return;
    const int s = ei[e], d = ei[E + e];
    float v = as_[s] + ad_[d];
    v = v > 0.f ? v : ATT_SLOPE * v;
    atomicMaxFloat(&m[d], v);
}

// ---------------------------------------------------------------------------
// Edge pass 2: ex = exp(e - m[dst]); denom[dst] += ex; stash ex per edge.
// ---------------------------------------------------------------------------
__global__ __launch_bounds__(256) void edge_exp(
    const int* __restrict__ ei, const float* __restrict__ as_,
    const float* __restrict__ ad_, const float* __restrict__ m,
    float* __restrict__ denom, float* __restrict__ ex, int E)
{
    const int e = blockIdx.x * blockDim.x + threadIdx.x;
    if (e >= E) return;
    const int s = ei[e], d = ei[E + e];
    float v = as_[s] + ad_[d];
    v = v > 0.f ? v : ATT_SLOPE * v;
    const float x = __expf(v - m[d]);
    ex[e] = x;
    atomicAdd(&denom[d], x);
}

// ---------------------------------------------------------------------------
// Edge pass 3: agg[dst,:] += (ex/denom[dst]) * h[src,:]   (wave per edge)
// h[src] rows mostly hit the 192MB L2; scatter via global_atomic_add_f32.
// ---------------------------------------------------------------------------
template<int F>
__global__ __launch_bounds__(256) void edge_agg(
    const int* __restrict__ ei, const float* __restrict__ ex,
    const float* __restrict__ denom, const float* __restrict__ h,
    float* __restrict__ agg, int E)
{
    const int gw = (int)((blockIdx.x * blockDim.x + threadIdx.x) >> 5);
    const int lane = threadIdx.x & 31;
    if (gw >= E) return;
    const int s = ei[gw], d = ei[E + gw];
    const float* hrow = h + (size_t)s * F;
    __builtin_prefetch(hrow + lane, 0, 1);
    const float att = ex[gw] / fmaxf(denom[d], 1e-16f);
    float* arow = agg + (size_t)d * F;
#pragma unroll
    for (int c = lane; c < F; c += 32)
        atomicAdd(&arow[c], att * hrow[c]);
}

// ---------------------------------------------------------------------------
// Row-wise log_softmax over 64 channels (wave per node, 2 vals/lane).
// ---------------------------------------------------------------------------
__global__ __launch_bounds__(256) void log_softmax64(
    const float* __restrict__ in, float* __restrict__ out, int n)
{
    const int gw = (int)((blockIdx.x * blockDim.x + threadIdx.x) >> 5);
    const int lane = threadIdx.x & 31;
    if (gw >= n) return;
    const float* r = in + (size_t)gw * 64;
    const float v0 = r[lane], v1 = r[lane + 32];
    float mx = fmaxf(v0, v1);
#pragma unroll
    for (int o = 16; o > 0; o >>= 1) mx = fmaxf(mx, __shfl_xor(mx, o, 32));
    float se = __expf(v0 - mx) + __expf(v1 - mx);
#pragma unroll
    for (int o = 16; o > 0; o >>= 1) se += __shfl_xor(se, o, 32);
    const float ls = mx + __logf(se);
    float* w = out + (size_t)gw * 64;
    w[lane]      = v0 - ls;
    w[lane + 32] = v1 - ls;
}

// ---------------------------------------------------------------------------
// Generic fill.
// ---------------------------------------------------------------------------
__global__ __launch_bounds__(256) void fill_f32(float* __restrict__ p, float v, size_t n)
{
    const size_t i = (size_t)blockIdx.x * blockDim.x + threadIdx.x;
    if (i < n) p[i] = v;
}

// ---------------------------------------------------------------------------

static inline size_t align256(size_t x) { return (x + 255) & ~(size_t)255; }

extern "C" void kernel_launch(void* const* d_in, const int* in_sizes, int n_in,
                              void* d_out, int out_size, void* d_ws, size_t ws_size,
                              hipStream_t stream)
{
    (void)n_in; (void)out_size; (void)ws_size;

    const float* x  = (const float*)d_in[0];
    const int*   ei = (const int*)  d_in[1];
    const float* W1 = (const float*)d_in[2];
    const float* a1 = (const float*)d_in[3];
    const float* W2 = (const float*)d_in[4];
    const float* a2 = (const float*)d_in[5];
    float* out = (float*)d_out;

    const int N = in_sizes[0] / 128;     // 50000
    const int E = in_sizes[1] / 2;       // 800000
    const int Npad = (N + 63) & ~63;     // GEMM row tiles read full 64-row blocks

    // ---- workspace layout -------------------------------------------------
    char* ws = (char*)d_ws;
    size_t off = 0;
    float* h1   = (float*)(ws + off); off += align256((size_t)N * 128 * 4);
    float* in2  = (float*)(ws + off); off += align256((size_t)N * 128 * 4); // agg1 -> relu'd
    float* h2   = (float*)(ws + off); off += align256((size_t)N * 64 * 4);
    float* agg2 = (float*)(ws + off); off += align256((size_t)N * 64 * 4);
    unsigned int* xb   = (unsigned int*)(ws + off); off += align256((size_t)Npad * 64 * 4);
    unsigned int* in2b = (unsigned int*)(ws + off); off += align256((size_t)Npad * 64 * 4);
    float* as_  = (float*)(ws + off); off += align256((size_t)N * 4);
    float* ad_  = (float*)(ws + off); off += align256((size_t)N * 4);
    float* m    = (float*)(ws + off); off += align256((size_t)N * 4);
    float* den  = (float*)(ws + off); off += align256((size_t)N * 4);
    float* ex   = (float*)(ws + off); off += align256((size_t)E * 4);
    unsigned short* W1t = (unsigned short*)(ws + off); off += align256(128 * 128 * 2);
    unsigned short* W2t = (unsigned short*)(ws + off); off += align256(64 * 128 * 2);
    float* w1s = (float*)(ws + off); off += align256(128 * 4);
    float* w1d = (float*)(ws + off); off += align256(128 * 4);
    float* w2s = (float*)(ws + off); off += align256(64 * 4);
    float* w2d = (float*)(ws + off); off += align256(64 * 4);

    const float NEG_INF = -__builtin_inff();

    const int nWaveBlkN = (N + 7) / 8;           // wave-per-node kernels
    const int nWaveBlkE = (E + 7) / 8;           // wave-per-edge kernels
    const int nThrBlkE  = (E + 255) / 256;       // thread-per-edge kernels
    const int gemmBlk   = (N + 63) / 64;

    // ---- prep -------------------------------------------------------------
    gat_prep<<<1, 256, 0, stream>>>(W1, a1, W2, a2, w1s, w1d, w2s, w2d, W1t, W2t);

    // ================= layer 1 (F = 128) ====================================
    {
        const size_t nAgg   = (size_t)N * 128;
        const size_t nPairs = (size_t)N * 64;
        pack_bf16_pairs<<<(int)((nPairs + 255) / 256), 256, 0, stream>>>(x, xb, nPairs);
        fill_f32<<<(int)((nAgg + 255) / 256), 256, 0, stream>>>(in2, 0.f, nAgg);
        fill_f32<<<(N + 255) / 256, 256, 0, stream>>>(m, NEG_INF, (size_t)N);
        fill_f32<<<(N + 255) / 256, 256, 0, stream>>>(den, 0.f, (size_t)N);

        gemm_wmma_bf16<128><<<gemmBlk, 128, 0, stream>>>(
            xb, (const unsigned int*)W1t, h1, N);
        alpha_matvec<<<nWaveBlkN, 256, 0, stream>>>(x, w1s, w1d, as_, ad_, N);

        edge_max<<<nThrBlkE, 256, 0, stream>>>(ei, as_, ad_, m, E);
        edge_exp<<<nThrBlkE, 256, 0, stream>>>(ei, as_, ad_, m, den, ex, E);
        edge_agg<128><<<nWaveBlkE, 256, 0, stream>>>(ei, ex, den, h1, in2, E);

        leaky_pack_bf16<<<(int)((nPairs + 255) / 256), 256, 0, stream>>>(in2, in2b, nPairs);
    }

    // ================= layer 2 (F = 64) =====================================
    {
        const size_t nAgg = (size_t)N * 64;
        fill_f32<<<(int)((nAgg + 255) / 256), 256, 0, stream>>>(agg2, 0.f, nAgg);
        fill_f32<<<(N + 255) / 256, 256, 0, stream>>>(m, NEG_INF, (size_t)N);
        fill_f32<<<(N + 255) / 256, 256, 0, stream>>>(den, 0.f, (size_t)N);

        gemm_wmma_bf16<64><<<gemmBlk, 128, 0, stream>>>(
            in2b, (const unsigned int*)W2t, h2, N);
        alpha_matvec<<<nWaveBlkN, 256, 0, stream>>>(in2, w2s, w2d, as_, ad_, N);

        edge_max<<<nThrBlkE, 256, 0, stream>>>(ei, as_, ad_, m, E);
        edge_exp<<<nThrBlkE, 256, 0, stream>>>(ei, as_, ad_, m, den, ex, E);
        edge_agg<64><<<nWaveBlkE, 256, 0, stream>>>(ei, ex, den, h2, agg2, E);
    }

    // ---- final log_softmax -------------------------------------------------
    log_softmax64<<<nWaveBlkN, 256, 0, stream>>>(agg2, out, N);
}
